// SelfAttention_5909874999361
// MI455X (gfx1250) — matrix operand order repaired and checked by reading, not compiled, not run on hardware
//
#include <hip/hip_runtime.h>
#include <hip/hip_bf16.h>

#define NTOT 4096            // H*W = 64*64
#define CCH  256
#define CQK  32
#define BATCH 8

typedef _Float16 h8   __attribute__((ext_vector_type(8)));
typedef _Float16 v16h __attribute__((ext_vector_type(16)));
typedef float    v8f  __attribute__((ext_vector_type(8)));
typedef float    v4f  __attribute__((ext_vector_type(4)));
typedef unsigned int u32x4 __attribute__((ext_vector_type(4)));
typedef int          i32x8 __attribute__((ext_vector_type(8)));
typedef int          i32x4 __attribute__((ext_vector_type(4)));

static __device__ inline v8f v8f_zero() {
    v8f z;
    #pragma unroll
    for (int i = 0; i < 8; ++i) z[i] = 0.f;
    return z;
}

static __device__ inline v16h cat16(h8 a, h8 b) {
    return __builtin_shufflevector(a, b, 0,1,2,3,4,5,6,7,8,9,10,11,12,13,14,15);
}

// ---- WMMA fragment loaders (ISA 7.12.2 layouts, wave32) -------------------
// A fragment: 16x32 f16, source row-major [16 rows][stride halves], K contiguous.
static __device__ inline v16h load_a_frag(const _Float16* base, int stride) {
    int lane = threadIdx.x & 31;
    int row  = lane & 15;
    int sel  = (lane >> 4) << 3;          // 0 or 8
    const _Float16* p = base + row * stride + sel;
    h8 lo = *(const h8*)(p);
    h8 hi = *(const h8*)(p + 16);
    return cat16(lo, hi);
}

// B fragment: 32x16 f16, source stored as B^T: [16 cols][stride halves], K contiguous.
static __device__ inline v16h load_b_frag(const _Float16* base, long stride) {
    int lane = threadIdx.x & 31;
    int col  = lane & 15;
    int kb   = (lane >> 4) << 3;          // 0 or 8
    const _Float16* p = base + (long)col * stride + kb;
    h8 lo = *(const h8*)(p);
    h8 hi = *(const h8*)(p + 16);
    return cat16(lo, hi);
}

static __device__ inline v8f wmma_f16(v16h a, v16h b, v8f c) {
    return __builtin_amdgcn_wmma_f32_16x16x32_f16(false, a, false, b, (short)0, c, false, false);
}

// ---- Kernel 0: convert all weights f32 -> f16 ------------------------------
__global__ __launch_bounds__(256) void cvt_w_kernel(
    const float* __restrict__ Wf, const float* __restrict__ Wg,
    const float* __restrict__ Wh, const float* __restrict__ Wv,
    _Float16* wf16, _Float16* wg16, _Float16* wh16, _Float16* wv16)
{
    int i = blockIdx.x * blockDim.x + threadIdx.x;
    if (i < CQK * CCH) { wf16[i] = (_Float16)Wf[i]; wg16[i] = (_Float16)Wg[i]; }
    if (i < CCH * CCH) { wh16[i] = (_Float16)Wh[i]; wv16[i] = (_Float16)Wv[i]; }
}

// ---- Kernel 1: projections f,g,h -------------------------------------------
// f,g -> fh/gh [B][N][32] (K-contiguous for attention A/B fragments)
// h   -> hh    [B][C][N]  (B^T layout for the P*V WMMA / TDM tile source)
#define XT_STRIDE 264   // 256 + 8 halves pad; rows stay 16B-aligned

__global__ __launch_bounds__(256) void proj_kernel(
    const float* __restrict__ x,
    const _Float16* __restrict__ wf16, const float* __restrict__ bf,
    const _Float16* __restrict__ wg16, const float* __restrict__ bg,
    const _Float16* __restrict__ wh16, const float* __restrict__ bh,
    _Float16* __restrict__ fh, _Float16* __restrict__ gh, _Float16* __restrict__ hh)
{
    __shared__ _Float16 xT[64 * XT_STRIDE];   // x tile transposed: [64 n][256 c] f16

    const int b  = blockIdx.y;
    const int n0 = blockIdx.x * 64;
    const float* xb = x + (size_t)b * CCH * NTOT;

    #pragma unroll 4
    for (int pass = 0; pass < 64; ++pass) {
        int c  = (threadIdx.x >> 6) + pass * 4;
        int nl = threadIdx.x & 63;
        xT[nl * XT_STRIDE + c] = (_Float16)xb[(size_t)c * NTOT + n0 + nl];
    }
    __syncthreads();

    const int wave = threadIdx.x >> 5;
    const int lane = threadIdx.x & 31;

    // 80 output tiles: 4 n-tiles x 20 o-tiles (2 f + 2 g + 16 h); 10 per wave
    for (int it = 0; it < 10; ++it) {
        int tile = it * 8 + wave;
        int nt = tile & 3;
        int ot = tile >> 2;
        const _Float16* wsel; const float* bsel; int o0; int which;
        if (ot < 2)      { wsel = wf16; bsel = bf; o0 = ot * 16;        which = 0; }
        else if (ot < 4) { wsel = wg16; bsel = bg; o0 = (ot - 2) * 16;  which = 1; }
        else             { wsel = wh16; bsel = bh; o0 = (ot - 4) * 16;  which = 2; }

        v8f acc = v8f_zero();
        #pragma unroll
        for (int kc = 0; kc < 8; ++kc) {
            v16h a  = load_a_frag(&xT[(nt * 16) * XT_STRIDE + kc * 32], XT_STRIDE);
            v16h bw = load_b_frag(wsel + (long)o0 * CCH + kc * 32, CCH);
            acc = wmma_f16(a, bw, acc);
        }

        int o     = o0 + (lane & 15);
        float bia = bsel[o];
        int nbase = n0 + nt * 16 + ((lane >> 4) << 3);
        if (which == 2) {
            h8 pk;
            #pragma unroll
            for (int r = 0; r < 8; ++r) pk[r] = (_Float16)(acc[r] + bia);
            *(h8*)(hh + ((size_t)b * CCH + o) * NTOT + nbase) = pk;
        } else {
            _Float16* dst = (which == 0 ? fh : gh) + (size_t)b * NTOT * CQK;
            #pragma unroll
            for (int r = 0; r < 8; ++r)
                dst[(size_t)(nbase + r) * CQK + o] = (_Float16)(acc[r] + bia);
        }
    }
}

// ---- Kernel 2: flash attention with TDM value staging ----------------------
// One block per (batch, 16-query tile). 8 waves; wave w owns c-slice [32w,32w+32).
// Per 128-key chunk: wave0 launches TENSOR_LOAD_TO_LDS of the [256c x 128j]
// value tile (overlapped with logits+softmax), then all waves run P*V WMMAs
// out of LDS. S/P never touch HBM; online softmax keeps running m/l.
#define SSTR 132      // S row stride (f32)
#define PSTR 136      // P row stride (halves), 16B-aligned rows
#define HSTR 136      // staged H row stride (halves) = 128 + TDM pad (4 DW)

// dynamic-LDS byte offsets (Hld must be 0: D#.lds_addr refers to it)
#define HLD_OFF   0
#define SLD_OFF   (CCH * HSTR * 2)              // 69632
#define P16_OFF   (SLD_OFF + 16 * SSTR * 4)     // 78080
#define MS_OFF    (P16_OFF + 16 * PSTR * 2)     // 82432
#define SMEM_SIZE (MS_OFF + 3 * 16 * 4)         // 82624

__global__ __launch_bounds__(256) void attn_kernel(
    const _Float16* __restrict__ fh, const _Float16* __restrict__ gh,
    const _Float16* __restrict__ hh, _Float16* __restrict__ ah)
{
    extern __shared__ char smem[];
    _Float16* Hld = (_Float16*)(smem + HLD_OFF);   // [256 c][136] staged values
    float*    Sld = (float*)   (smem + SLD_OFF);
    _Float16* P16 = (_Float16*)(smem + P16_OFF);
    float*    m_s = (float*)   (smem + MS_OFF);
    float*    l_s = m_s + 16;
    float*    alpha_s = m_s + 32;

    const int b  = blockIdx.y;
    const int i0 = blockIdx.x * 16;
    const int wave = threadIdx.x >> 5;
    const int lane = threadIdx.x & 31;

    const _Float16* fb = fh + (size_t)b * NTOT * CQK;
    const _Float16* gb = gh + (size_t)b * NTOT * CQK;
    const _Float16* hb = hh + (size_t)b * CCH * NTOT;

    // query fragment: rows i0..i0+15, full K=32 QK channels
    v16h afrag = load_a_frag(fb + (size_t)i0 * CQK, CQK);

    v8f o_acc0 = v8f_zero(), o_acc1 = v8f_zero();
    const int c0 = wave * 32;

    if (threadIdx.x < 16) { m_s[threadIdx.x] = -1e30f; l_s[threadIdx.x] = 0.f; }
    __syncthreads();

    for (int j0 = 0; j0 < NTOT; j0 += 128) {
        // --- TDM: DMA value tile hh[b][0:256][j0:j0+128] -> LDS (wave 0 only).
        // D# group0: count=1, lds_addr=0, 57b global addr of tile start, type=2.
        // D# group1: data_size=2B, pad 4 DW per 64 DW (row stride 136 halves),
        //            tensor 4096 x 256 (stride 4096), tile 128 x 256.
        if (wave == 0) {
            unsigned long long ga =
                (unsigned long long)(uintptr_t)hb + (unsigned long long)j0 * 2ull;
            u32x4 g0;
            g0[0] = 1u;                                   // count = 1
            g0[1] = 0u;                                   // lds_addr = HLD_OFF
            g0[2] = (unsigned)(ga & 0xffffffffull);
            g0[3] = (unsigned)((ga >> 32) & 0x1ffffffull) | (2u << 30); // type=2
            i32x8 g1;
            g1[0] = (int)((1u << 16)     // data_size = 2 bytes
                        | (1u << 20)     // pad_enable
                        | (5u << 22)     // pad_interval: 64 DWORDs (256B row)
                        | (3u << 25));   // pad_amount: 4 DWORDs (16B)
            g1[1] = (int)(4096u << 16);  // tensor_dim0[15:0] (j length)
            g1[2] = (int)(256u << 16);   // tensor_dim0[31:16]=0 | tensor_dim1[15:0]
            g1[3] = (int)(128u << 16);   // tensor_dim1[31:16]=0 | tile_dim0=128
            g1[4] = 256;                 // tile_dim1=256 | tile_dim2=0
            g1[5] = 4096;                // tensor_dim0_stride[31:0]
            g1[6] = 0;                   // stride0[47:32] | stride1[15:0]
            g1[7] = 0;                   // stride1[47:16]
            i32x4 z4;
            z4[0] = 0; z4[1] = 0; z4[2] = 0; z4[3] = 0;   // 2D tile: groups 2/3 unused
            i32x8 z8;
            #pragma unroll
            for (int q = 0; q < 8; ++q) z8[q] = 0;
            __builtin_amdgcn_tensor_load_to_lds(g0, g1, z4, z4, z8, 0);
        }

        // --- logits (overlaps the DMA): wave w computes S tile for keys j0+16w
        v16h gfrag = load_b_frag(gb + (size_t)(j0 + wave * 16) * CQK, CQK);
        v8f  s = wmma_f16(afrag, gfrag, v8f_zero());
        {
            int col = lane & 15, rb = (lane >> 4) << 3;
            #pragma unroll
            for (int r = 0; r < 8; ++r) Sld[(rb + r) * SSTR + wave * 16 + col] = s[r];
        }
        __syncthreads();

        // --- online softmax: 16 threads per row, 8 elements each
        {
            int row = threadIdx.x >> 4;
            int sub = threadIdx.x & 15;
            float v[8];
            float mx = -1e30f;
            #pragma unroll
            for (int e = 0; e < 8; ++e) {
                v[e] = Sld[row * SSTR + sub * 8 + e];
                mx = fmaxf(mx, v[e]);
            }
            #pragma unroll
            for (int off = 8; off; off >>= 1) mx = fmaxf(mx, __shfl_xor(mx, off, 16));
            float m_old = m_s[row];
            float m_new = fmaxf(m_old, mx);
            float psum = 0.f;
            #pragma unroll
            for (int e = 0; e < 8; ++e) {
                float p = __expf(v[e] - m_new);
                psum += p;
                P16[row * PSTR + sub * 8 + e] = (_Float16)p;
            }
            #pragma unroll
            for (int off = 8; off; off >>= 1) psum += __shfl_xor(psum, off, 16);
            if (sub == 0) {
                float al = __expf(m_old - m_new);
                alpha_s[row] = al;
                l_s[row] = l_s[row] * al + psum;
                m_s[row] = m_new;
            }
        }
        __syncthreads();

        // --- rescale running O by alpha (per query row)
        {
            int rb = (lane >> 4) << 3;
            #pragma unroll
            for (int r = 0; r < 8; ++r) {
                float al = alpha_s[rb + r];
                o_acc0[r] *= al;
                o_acc1[r] *= al;
            }
        }

        // --- wait for the value-tile DMA, make it visible to all waves
        if (wave == 0) __builtin_amdgcn_s_wait_tensorcnt(0);
        __syncthreads();

        // --- O += P * H^T from LDS (wave's 32-channel slice)
        #pragma unroll
        for (int jj = 0; jj < 4; ++jj) {
            v16h pa = load_a_frag(&P16[jj * 32], PSTR);
            v16h hb0 = load_b_frag(&Hld[(c0)      * HSTR + jj * 32], HSTR);
            o_acc0 = wmma_f16(pa, hb0, o_acc0);
            v16h hb1 = load_b_frag(&Hld[(c0 + 16) * HSTR + jj * 32], HSTR);
            o_acc1 = wmma_f16(pa, hb1, o_acc1);
        }
        __syncthreads();   // release Hld/Sld/P16 before next chunk
    }

    // --- epilogue: divide by softmax denom, store att_h as [b][n][c] f16
    {
        int col = lane & 15, rb = (lane >> 4) << 3;
        #pragma unroll
        for (int r = 0; r < 8; ++r) {
            float inv = 1.f / l_s[rb + r];
            int n = i0 + rb + r;
            ah[((size_t)b * NTOT + n) * CCH + c0 + col]      = (_Float16)(o_acc0[r] * inv);
            ah[((size_t)b * NTOT + n) * CCH + c0 + 16 + col] = (_Float16)(o_acc1[r] * inv);
        }
    }
}

// ---- Kernel 3: out = x + gamma * (Wv * att_h + bv) -------------------------
__global__ __launch_bounds__(256) void outproj_kernel(
    const float* __restrict__ x, const _Float16* __restrict__ ah,
    const _Float16* __restrict__ wv16, const float* __restrict__ bv,
    const float* __restrict__ gamma, float* __restrict__ out)
{
    const int b  = blockIdx.y;
    const int n0 = blockIdx.x * 16;
    const int wave = threadIdx.x >> 5;
    const int lane = threadIdx.x & 31;
    const _Float16* ab = ah + (size_t)b * NTOT * CCH;
    const float g = gamma[0];

    v8f acc0 = v8f_zero(), acc1 = v8f_zero();
    #pragma unroll
    for (int kc = 0; kc < 8; ++kc) {
        v16h a  = load_a_frag(ab + (size_t)n0 * CCH + kc * 32, CCH);
        v16h b0 = load_b_frag(wv16 + (long)(wave * 32)      * CCH + kc * 32, CCH);
        acc0 = wmma_f16(a, b0, acc0);
        v16h b1 = load_b_frag(wv16 + (long)(wave * 32 + 16) * CCH + kc * 32, CCH);
        acc1 = wmma_f16(a, b1, acc1);
    }

    int col = lane & 15, rb = (lane >> 4) << 3;
    #pragma unroll
    for (int t = 0; t < 2; ++t) {
        v8f acc = t ? acc1 : acc0;
        int o = wave * 32 + t * 16 + col;
        float bia = bv[o];
        size_t base = ((size_t)b * CCH + o) * NTOT + n0 + rb;  // 8 contiguous n
        v4f x0 = *(const v4f*)(x + base);
        v4f x1 = *(const v4f*)(x + base + 4);
        v4f r0, r1;
        #pragma unroll
        for (int r = 0; r < 4; ++r) {
            r0[r] = x0[r] + g * (acc[r]     + bia);
            r1[r] = x1[r] + g * (acc[r + 4] + bia);
        }
        *(v4f*)(out + base)     = r0;
        *(v4f*)(out + base + 4) = r1;
    }
}

// ---- host launch -----------------------------------------------------------
extern "C" void kernel_launch(void* const* d_in, const int* in_sizes, int n_in,
                              void* d_out, int out_size, void* d_ws, size_t ws_size,
                              hipStream_t stream) {
    const float* x     = (const float*)d_in[0];
    const float* Wf    = (const float*)d_in[1];
    const float* bf    = (const float*)d_in[2];
    const float* Wg    = (const float*)d_in[3];
    const float* bg    = (const float*)d_in[4];
    const float* Wh    = (const float*)d_in[5];
    const float* bh    = (const float*)d_in[6];
    const float* Wv    = (const float*)d_in[7];
    const float* bv    = (const float*)d_in[8];
    const float* gamma = (const float*)d_in[9];
    float* out = (float*)d_out;

    char* ws = (char*)d_ws;
    size_t off = 0;
    auto carve = [&](size_t bytes) -> char* {
        char* p = ws + off;
        off += (bytes + 255) & ~(size_t)255;
        return p;
    };
    _Float16* wf16 = (_Float16*)carve((size_t)CQK * CCH * 2);
    _Float16* wg16 = (_Float16*)carve((size_t)CQK * CCH * 2);
    _Float16* wh16 = (_Float16*)carve((size_t)CCH * CCH * 2);
    _Float16* wv16 = (_Float16*)carve((size_t)CCH * CCH * 2);
    _Float16* fh   = (_Float16*)carve((size_t)BATCH * NTOT * CQK * 2);
    _Float16* gh   = (_Float16*)carve((size_t)BATCH * NTOT * CQK * 2);
    _Float16* hh   = (_Float16*)carve((size_t)BATCH * CCH * NTOT * 2);
    _Float16* ah   = (_Float16*)carve((size_t)BATCH * NTOT * CCH * 2);

    cvt_w_kernel<<<dim3(256), 256, 0, stream>>>(Wf, Wg, Wh, Wv, wf16, wg16, wh16, wv16);
    proj_kernel<<<dim3(NTOT / 64, BATCH), 256, 0, stream>>>(
        x, wf16, bf, wg16, bg, wh16, bh, fh, gh, hh);
    attn_kernel<<<dim3(NTOT / 16, BATCH), 256, SMEM_SIZE, stream>>>(fh, gh, hh, ah);
    outproj_kernel<<<dim3(NTOT / 16, BATCH), 256, 0, stream>>>(x, ah, wv16, bv, gamma, out);
}
